// DifferentialQuadraticSplineStack_16106127360534
// MI455X (gfx1250) — compile-verified, matrix operand order checked by reading, not run on hardware
//
#include <hip/hip_runtime.h>
#include <hip/hip_bf16.h>

#define NR 8
#define NG 1000
#define SUMH 224
#define SUMW 221

typedef __attribute__((ext_vector_type(16))) _Float16 v16h;
typedef __attribute__((ext_vector_type(8)))  float    v8f;
typedef __attribute__((ext_vector_type(4)))  float    fvec4;

// ======================= phase 1: spline table build =======================

__device__ __forceinline__ float brMax128(float v, float* s) {
  int t = threadIdx.x; s[t] = v; __syncthreads();
  #pragma unroll
  for (int o = 64; o > 0; o >>= 1) { if (t < o) s[t] = fmaxf(s[t], s[t + o]); __syncthreads(); }
  float r = s[0]; __syncthreads(); return r;
}
__device__ __forceinline__ float brSum128(float v, float* s) {
  int t = threadIdx.x; s[t] = v; __syncthreads();
  #pragma unroll
  for (int o = 64; o > 0; o >>= 1) { if (t < o) s[t] = s[t] + s[t + o]; __syncthreads(); }
  float r = s[0]; __syncthreads(); return r;
}

__global__ void k_init_gs(float* gs) {
  int i = blockIdx.x * blockDim.x + threadIdx.x;
  if (i < NR * NG) gs[i] = 1.0f / (float)NG;
}

// One block (128 threads) per (r,g): softmax widths, exp heights, per-gene
// trapezoid mass P (without genespacing factor). Stores e and softmax-w for pass 3.
__global__ void k_gene(const float* __restrict__ UH, const float* __restrict__ UW,
                       const float* __restrict__ MDG,
                       float* __restrict__ Ebuf, float* __restrict__ SWbuf,
                       float* __restrict__ Pbuf,
                       int n, int hoff, int woff)
{
  __shared__ float sh[128];
  __shared__ float eSh[129];
  int b = blockIdx.x, r = b / NG, g = b % NG, t = threadIdx.x;
  const float* uwp = UW + (size_t)g * SUMW + woff;
  const float* uhp = UH + (size_t)g * SUMH + hoff;
  const float* dhp = MDG + (size_t)b * SUMH + hoff;   // b == r*NG + g

  float uw = (t < n - 1) ? uwp[t] : -1e30f;
  float m  = brMax128(uw, sh);
  float ew = (t < n - 1) ? expf(uw - m) : 0.f;
  float se = brSum128(ew, sh);
  float sw = (t < n - 1) ? ew / se : 0.f;

  float e = (t < n) ? expf(uhp[t] + dhp[t]) : 0.f;
  eSh[t] = e; if (t == 0) eSh[128] = 0.f;
  __syncthreads();
  float e1 = eSh[t + 1];
  float d  = (t < n - 1) ? 0.5f * (e + e1) * sw : 0.f;
  float P  = brSum128(d, sh);

  Ebuf[(size_t)b * 128 + t] = e;
  if (r == 0) SWbuf[(size_t)g * 128 + t] = sw;
  if (t == 0) Pbuf[b] = P;
}

// One block per r: exclusive scans over genes (f64 to avoid prefix cancellation
// error that gets amplified by 1/binwidth downstream), per-r area, next genespacing.
__global__ void k_scan_genes(const float* __restrict__ Pbuf, const float* __restrict__ gsIn,
                             double* __restrict__ locPref, double* __restrict__ cdfPref,
                             float* __restrict__ gsOut, double* __restrict__ areaOut)
{
  __shared__ double sx[1024];
  __shared__ double sy[1024];
  int r = blockIdx.x, g = threadIdx.x;
  double gs = (g < NG) ? (double)gsIn[r * NG + g] : 0.0;
  double Pv = (g < NG) ? (double)Pbuf[r * NG + g] : 0.0;
  double px = gs, py = gs * Pv;
  double ix = px, iy = py;
  sx[g] = ix; sy[g] = iy; __syncthreads();
  for (int o = 1; o < 1024; o <<= 1) {
    double ax = 0.0, ay = 0.0;
    if (g >= o) { ax = sx[g - o]; ay = sy[g - o]; }
    __syncthreads();
    ix += ax; iy += ay; sx[g] = ix; sy[g] = iy; __syncthreads();
  }
  double area = sy[1023];
  if (g < NG) {
    locPref[r * NG + g] = ix - px;
    cdfPref[r * NG + g] = iy - py;
    float gsn = (float)(py / area);
    if (g == NG - 1) gsn = (float)(1.0 - (iy - py) / area);  // cdf[...,-1]=1.0 override
    gsOut[r * NG + g] = gsn;
  }
  if (g == 0) areaOut[r] = area;
}

// One block per (r,g): local scans + global prefixes -> final f32 tables.
__global__ void k_tables(const float* __restrict__ Ebuf, const float* __restrict__ SWbuf,
                         const float* __restrict__ gsCur,
                         const double* __restrict__ locPref, const double* __restrict__ cdfPref,
                         const double* __restrict__ areaBuf,
                         float* __restrict__ Wt, float* __restrict__ Ht,
                         float* __restrict__ Ct, float* __restrict__ Lt, int n)
{
  __shared__ float eSh[129];
  __shared__ float sx[128];
  __shared__ float sy[128];
  int b = blockIdx.x, r = b / NG, g = b % NG, t = threadIdx.x;
  double area = areaBuf[r];
  float  gs = gsCur[r * NG + g];
  double lp = locPref[r * NG + g];
  double cp = cdfPref[r * NG + g];

  float e  = (t < n)     ? Ebuf[(size_t)b * 128 + t]  : 0.f;
  float sw = (t < n - 1) ? SWbuf[(size_t)g * 128 + t] : 0.f;
  float w  = sw * gs;

  eSh[t] = e; if (t == 0) eSh[128] = 0.f;
  __syncthreads();
  float e1 = eSh[t + 1];
  float d  = (t < n - 1) ? 0.5f * (e + e1) * w : 0.f;

  float ix = w, iy = d;
  sx[t] = ix; sy[t] = iy; __syncthreads();
  #pragma unroll
  for (int o = 1; o < 128; o <<= 1) {
    float ax = 0.f, ay = 0.f;
    if (t >= o) { ax = sx[t - o]; ay = sy[t - o]; }
    __syncthreads();
    ix += ax; iy += ay; sx[t] = ix; sy[t] = iy; __syncthreads();
  }
  if (t < n) {
    size_t idx = (size_t)r * ((size_t)NG * n) + (size_t)g * n + t;
    bool last = (g == NG - 1) && (t == n - 1);
    Wt[idx] = w;                                   // pad slot (t==n-1) is exactly 0
    Ht[idx] = (float)((double)e / area);
    Lt[idx] = last ? 1.0f : (float)(lp + (double)(ix - w));
    Ct[idx] = last ? 1.0f : (float)((cp + (double)(iy - d)) / area);
  }
}

// ======================= phase 2: WMMA-counted spline eval =======================

__device__ __forceinline__ _Float16 ind1(float v, float x) {
  return (_Float16)((v < x) ? 1.0f : 0.0f);
}

// Build one lane's A-fragment half: 16 indicator bits for bins [c0..c0+7] and
// [c0+16..c0+23] of this row (contiguous 32B runs -> float4 loads). With B=ones
// the count is invariant to any within-row K permutation, so only the
// documented M=lane%16 row mapping matters.
__device__ __forceinline__ v16h indicators16(const float* __restrict__ p, float x) {
  const fvec4* q = (const fvec4*)p;
  fvec4 a0 = q[0], a1 = q[1], b0 = q[4], b1 = q[5];
  v16h a;
  a[0]  = ind1(a0.x, x); a[1]  = ind1(a0.y, x); a[2]  = ind1(a0.z, x); a[3]  = ind1(a0.w, x);
  a[4]  = ind1(a1.x, x); a[5]  = ind1(a1.y, x); a[6]  = ind1(a1.z, x); a[7]  = ind1(a1.w, x);
  a[8]  = ind1(b0.x, x); a[9]  = ind1(b0.y, x); a[10] = ind1(b0.z, x); a[11] = ind1(b0.w, x);
  a[12] = ind1(b1.x, x); a[13] = ind1(b1.y, x); a[14] = ind1(b1.z, x); a[15] = ind1(b1.w, x);
  return a;
}

template<int N>
__device__ __forceinline__ void layer_step(float& x, float& lad,
    const float* __restrict__ Wt, const float* __restrict__ Ht,
    const float* __restrict__ Ct, const float* __restrict__ Lt,
    int gene, int refl, int lane)
{
  int rowOff = refl * (NG * N) + gene * N;   // row base in (R, G*N) tables
  int M    = lane & 15;
  int half = lane >> 4;

  // WMMA#1 rows 0-7 host cuts 0-7, rows 8-15 host cuts 16-23; WMMA#2 = +8.
  int o1 = M + ((M & 8) ? 8 : 0);
  int o2 = o1 + 8;
  float x1  = __shfl(x, o1, 32);      int rb1 = __shfl(rowOff, o1, 32);
  float x2  = __shfl(x, o2, 32);      int rb2 = __shfl(rowOff, o2, 32);

  v16h ones;
  #pragma unroll
  for (int i = 0; i < 16; i++) ones[i] = (_Float16)1.0f;
  v8f c1 = {};
  v8f c2 = {};

  #pragma unroll
  for (int kk = 0; kk < N; kk += 32) {
    int c0 = kk + half * 8;
    v16h a1 = indicators16(Lt + rb1 + c0, x1);
    v16h a2 = indicators16(Lt + rb2 + c0, x2);
    c1 = __builtin_amdgcn_wmma_f32_16x16x32_f16(false, a1, false, ones, (short)0, c1, false, false);
    c2 = __builtin_amdgcn_wmma_f32_16x16x32_f16(false, a2, false, ones, (short)0, c2, false, false);
  }

  // Extract this lane's count from the documented C layout (M = j + 8*half).
  v8f cs;
  if (lane & 8) cs = c2; else cs = c1;
  float cnt = cs[0];
  #pragma unroll
  for (int j = 1; j < 8; j++) if ((lane & 7) == j) cnt = cs[j];

  int ss = (int)cnt;                       // exact integer: 0/1 products in f16, <=128 in f32
  int bi = ss - 1;
  bi = bi < 0 ? 0 : (bi > N - 2 ? N - 2 : bi);
  int idx = rowOff + bi;

  float iw = Wt[idx];
  float hl = Ht[idx];
  float hr = Ht[idx + 1];
  float cl = Ct[idx];
  float ll = Lt[idx];
  float alpha = (x - ll) / iw;
  float outv  = (0.5f * (hr - hl) * iw * alpha + hl * iw) * alpha + cl;
  x = fminf(fmaxf(outv, 0.0f), 1.0f);
  lad += logf(hl + alpha * (hr - hl));
}

__global__ void k_forward(const float* __restrict__ cutPos,
                          const int* __restrict__ geneIx, const int* __restrict__ reflIx,
                          const float* __restrict__ W0, const float* __restrict__ H0,
                          const float* __restrict__ C0, const float* __restrict__ L0,
                          const float* __restrict__ W1, const float* __restrict__ H1,
                          const float* __restrict__ C1, const float* __restrict__ L1,
                          const float* __restrict__ W2, const float* __restrict__ H2,
                          const float* __restrict__ C2, const float* __restrict__ L2,
                          float* __restrict__ out, int ncuts)
{
  int tid  = blockIdx.x * blockDim.x + threadIdx.x;
  int lane = threadIdx.x & 31;
  bool valid = tid < ncuts;
  int cid = valid ? tid : 0;           // keep EXEC full for WMMA; clamp index only
  float x = cutPos[cid];
  int g   = geneIx[cid];
  int r   = reflIx[cid];
  float lad = 0.f;

  layer_step<128>(x, lad, W0, H0, C0, L0, g, r, lane);
  layer_step< 64>(x, lad, W1, H1, C1, L1, g, r, lane);
  layer_step< 32>(x, lad, W2, H2, C2, L2, g, r, lane);

  if (valid) {
    out[tid]         = x;
    out[ncuts + tid] = lad;
  }
}

// ======================= launch =======================

extern "C" void kernel_launch(void* const* d_in, const int* in_sizes, int n_in,
                              void* d_out, int out_size, void* d_ws, size_t ws_size,
                              hipStream_t stream)
{
  (void)n_in; (void)out_size; (void)ws_size;
  const float* cutPos = (const float*)d_in[0];
  const float* MDG    = (const float*)d_in[1];
  const float* UH     = (const float*)d_in[2];
  const float* UW     = (const float*)d_in[3];
  const int*   geneIx = (const int*)d_in[5];
  const int*   reflIx = (const int*)d_in[6];
  float* out = (float*)d_out;
  const int ncuts = in_sizes[0];

  char* p = (char*)d_ws;
  auto carve = [&](size_t bytes) -> void* {
    void* q = (void*)p;
    p += (bytes + 255) & ~(size_t)255;
    return q;
  };

  const int NB[3] = {128, 64, 32};
  const int HO[3] = {0, 128, 192};
  const int WO[3] = {0, 127, 190};

  float *Wt[3], *Ht[3], *Ct[3], *Lt[3];
  for (int l = 0; l < 3; l++) {
    size_t tb = (size_t)NR * NG * NB[l] * sizeof(float);
    Wt[l] = (float*)carve(tb);
    Ht[l] = (float*)carve(tb);
    Ct[l] = (float*)carve(tb);
    Lt[l] = (float*)carve(tb);
  }
  float*  Ebuf    = (float*) carve((size_t)NR * NG * 128 * sizeof(float));
  float*  SWbuf   = (float*) carve((size_t)NG * 128 * sizeof(float));
  float*  Pbuf    = (float*) carve((size_t)NR * NG * sizeof(float));
  double* locPref = (double*)carve((size_t)NR * NG * sizeof(double));
  double* cdfPref = (double*)carve((size_t)NR * NG * sizeof(double));
  float*  gsA     = (float*) carve((size_t)NR * NG * sizeof(float));
  float*  gsB     = (float*) carve((size_t)NR * NG * sizeof(float));
  double* areaBuf = (double*)carve((size_t)NR * sizeof(double));

  k_init_gs<<<(NR * NG + 255) / 256, 256, 0, stream>>>(gsA);

  float* gsCur = gsA;
  float* gsNext = gsB;
  for (int l = 0; l < 3; l++) {
    k_gene<<<NR * NG, 128, 0, stream>>>(UH, UW, MDG, Ebuf, SWbuf, Pbuf, NB[l], HO[l], WO[l]);
    k_scan_genes<<<NR, 1024, 0, stream>>>(Pbuf, gsCur, locPref, cdfPref, gsNext, areaBuf);
    k_tables<<<NR * NG, 128, 0, stream>>>(Ebuf, SWbuf, gsCur, locPref, cdfPref, areaBuf,
                                          Wt[l], Ht[l], Ct[l], Lt[l], NB[l]);
    float* tmp = gsCur; gsCur = gsNext; gsNext = tmp;
  }

  k_forward<<<(ncuts + 255) / 256, 256, 0, stream>>>(
      cutPos, geneIx, reflIx,
      Wt[0], Ht[0], Ct[0], Lt[0],
      Wt[1], Ht[1], Ct[1], Lt[1],
      Wt[2], Ht[2], Ct[2], Lt[2],
      out, ncuts);
}
